// FontMatchModel_52570399703481
// MI455X (gfx1250) — matrix operand order repaired and checked by reading, not compiled, not run on hardware
//
#include <hip/hip_runtime.h>

// ---------------------------------------------------------------------------
// Types for CDNA5 WMMA bf16 (16x16x32, fp32 accumulate)
// ---------------------------------------------------------------------------
typedef __attribute__((ext_vector_type(16))) __bf16 v16bf;
typedef __attribute__((ext_vector_type(8)))  float  v8f;
typedef __attribute__((ext_vector_type(4)))  unsigned int u32x4;

union FragAB { v16bf bf; u32x4 u[2]; };

__device__ __forceinline__ v8f wmma_bf16(const FragAB& a, const FragAB& b, v8f c) {
  return __builtin_amdgcn_wmma_f32_16x16x32_bf16(false, a.bf, false, b.bf,
                                                 (short)0, c, false, false);
}

__device__ __forceinline__ unsigned short f2bf(float x) {
  unsigned int u = __float_as_uint(x);
  unsigned int r = u + 0x7fffu + ((u >> 16) & 1u);   // round-to-nearest-even
  return (unsigned short)(r >> 16);
}

// ---------------------------------------------------------------------------
// fp32 -> bf16 converters
// ---------------------------------------------------------------------------
__global__ __launch_bounds__(256) void k_convert(const float* __restrict__ s,
                                                 unsigned short* __restrict__ d, int n) {
  int i = blockIdx.x * 256 + threadIdx.x;
  if (i < n) d[i] = f2bf(s[i]);
}

__global__ __launch_bounds__(256) void k_convert_pad(const float* __restrict__ s,
                                                     unsigned short* __restrict__ d,
                                                     int rows, int cols, int drows) {
  int i = blockIdx.x * 256 + threadIdx.x;
  if (i >= drows * cols) return;
  int r = i / cols;
  d[i] = (r < rows) ? f2bf(s[i]) : (unsigned short)0;
}

// ---------------------------------------------------------------------------
// Generic bf16 WMMA GEMM:  C[M,N] = act(BN(A[M,K] @ W[N,K]^T + bias))
// Wave tile 32x32 (2x2 frags), WG 8 waves -> 64(M) x 128(N).
// ---------------------------------------------------------------------------
__global__ __launch_bounds__(256) void k_gemm_bf16(
    const unsigned short* __restrict__ A, int lda,
    const unsigned short* __restrict__ W, int ldw,
    const float* __restrict__ bias,
    const float* __restrict__ gam, const float* __restrict__ bet,
    const float* __restrict__ mu,  const float* __restrict__ var,
    float* __restrict__ outF, unsigned short* __restrict__ outB, int ldo,
    int M, int Npad, int K, int n_store, int relu)
{
  const int lane = threadIdx.x & 31;
  const int wave = threadIdx.x >> 5;
  const int m0 = blockIdx.y * 64 + (wave >> 2) * 32;
  const int n0 = blockIdx.x * 128 + (wave & 3) * 32;
  if (m0 >= M || n0 >= Npad) return;

  const int mrow = lane & 15;
  const int ka = (lane < 16) ? 0 : 8;    // A: K {0-7,16-23} / {8-15,24-31}
  const int kb = (lane < 16) ? 0 : 8;    // B: K {0-7,16-23} / {8-15,24-31}

  const unsigned short* a0 = A + (size_t)(m0 + mrow) * lda + ka;
  const unsigned short* a1 = A + (size_t)(m0 + 16 + mrow) * lda + ka;
  const unsigned short* b0 = W + (size_t)(n0 + mrow) * ldw + kb;
  const unsigned short* b1 = W + (size_t)(n0 + 16 + mrow) * ldw + kb;

  v8f acc00 = {0.f,0.f,0.f,0.f,0.f,0.f,0.f,0.f};
  v8f acc01 = acc00, acc10 = acc00, acc11 = acc00;

  for (int k = 0; k < K; k += 32) {
    FragAB fa0, fa1, fb0, fb1;
    fa0.u[0] = *(const u32x4*)(a0 + k);      fa0.u[1] = *(const u32x4*)(a0 + k + 16);
    fa1.u[0] = *(const u32x4*)(a1 + k);      fa1.u[1] = *(const u32x4*)(a1 + k + 16);
    fb0.u[0] = *(const u32x4*)(b0 + k);      fb0.u[1] = *(const u32x4*)(b0 + k + 16);
    fb1.u[0] = *(const u32x4*)(b1 + k);      fb1.u[1] = *(const u32x4*)(b1 + k + 16);
    acc00 = wmma_bf16(fa0, fb0, acc00);
    acc01 = wmma_bf16(fa0, fb1, acc01);
    acc10 = wmma_bf16(fa1, fb0, acc10);
    acc11 = wmma_bf16(fa1, fb1, acc11);
  }

  const int ncol  = n0 + (lane & 15);
  const int mbase = m0 + (lane >> 4) * 8;

  v8f* accs[4] = { &acc00, &acc01, &acc10, &acc11 };
  for (int t = 0; t < 4; ++t) {
    int mi = t >> 1, ni = t & 1;
    int n = ncol + ni * 16;
    if (n >= n_store) continue;
    float bv = bias ? bias[n] : 0.f;
    float sc = 1.f, sh = 0.f;
    if (gam) { sc = gam[n] * rsqrtf(var[n] + 1e-5f); sh = bet[n] - mu[n] * sc; }
    v8f a = *accs[t];
    for (int j = 0; j < 8; ++j) {
      float v = a[j] + bv;
      if (gam)  v = v * sc + sh;
      if (relu) v = fmaxf(v, 0.f);
      size_t idx = (size_t)(mbase + mi * 16 + j) * ldo + n;
      if (outF) outF[idx] = v;
      if (outB) outB[idx] = f2bf(v);
    }
  }
}

// ---------------------------------------------------------------------------
// GRU recurrence. One WG handles 16 batch rows (batch rows are independent).
// Per step: gh = h_bf @ Whh^T + bhh via WMMA (A from LDS), then gate math.
// Emits relu(h_t) as bf16 ready to be the next GEMM's A operand.
// ---------------------------------------------------------------------------
__global__ __launch_bounds__(256) void k_gru(
    const float* __restrict__ xp,            // [B*S, 3H]
    const unsigned short* __restrict__ Whh,  // [3H, H] bf16
    const float* __restrict__ bhh,           // [3H]
    unsigned short* __restrict__ out_relu,   // [B*S, H] bf16
    int H, int S)
{
  extern __shared__ char smem[];
  unsigned short* h_bf = (unsigned short*)smem;                 // 16*H bf16
  float* h_f = (float*)(smem + (size_t)16 * H * 2);             // 16*H f32
  float* gh  = (float*)(smem + (size_t)16 * H * 2 + (size_t)16 * H * 4); // 16*3H f32

  const int r0   = blockIdx.x * 16;
  const int lane = threadIdx.x & 31;
  const int wave = threadIdx.x >> 5;
  const int H3   = 3 * H;

  for (int i = threadIdx.x; i < 16 * H; i += 256) { h_bf[i] = 0; h_f[i] = 0.f; }
  __syncthreads();

  const int tiles_per_wave = (H3 / 16) / 8;   // 3 (H=128) or 6 (H=256)
  const int mrow = lane & 15;
  const int ka = (lane < 16) ? 0 : 8;
  const int kb = (lane < 16) ? 0 : 8;
  const unsigned short* ap = h_bf + (size_t)mrow * H + ka;

  for (int t = 0; t < S; ++t) {
    // --- GEMM phase: WMMA tiles over N = 3H, K = H, M = 16
    for (int tt = 0; tt < tiles_per_wave; ++tt) {
      int n0 = (wave * tiles_per_wave + tt) * 16;
      const unsigned short* bp = Whh + (size_t)(n0 + mrow) * H + kb;
      v8f acc = {0.f,0.f,0.f,0.f,0.f,0.f,0.f,0.f};
      for (int k = 0; k < H; k += 32) {
        FragAB fa, fb;
        fa.u[0] = *(const u32x4*)(ap + k);  fa.u[1] = *(const u32x4*)(ap + k + 16);
        fb.u[0] = *(const u32x4*)(bp + k);  fb.u[1] = *(const u32x4*)(bp + k + 16);
        acc = wmma_bf16(fa, fb, acc);
      }
      int n  = n0 + mrow;
      int mb = (lane >> 4) * 8;
      float bv = bhh[n];
      for (int j = 0; j < 8; ++j) gh[(size_t)(mb + j) * H3 + n] = acc[j] + bv;
    }
    __syncthreads();

    // --- gate phase: r, z, n gates, PyTorch order
    for (int i = threadIdx.x; i < 16 * H; i += 256) {
      int m = i / H, j = i % H;
      size_t row = ((size_t)(r0 + m) * S + t);
      const float* x = xp + row * H3;
      float xr = x[j], xz = x[H + j], xn = x[2 * H + j];
      const float* ghr = gh + (size_t)m * H3;
      float hr = ghr[j], hz = ghr[H + j], hn = ghr[2 * H + j];
      float r  = 1.f / (1.f + __expf(-(xr + hr)));
      float z  = 1.f / (1.f + __expf(-(xz + hz)));
      float ng = tanhf(xn + r * hn);
      float hnew = (1.f - z) * ng + z * h_f[i];
      h_f[i]  = hnew;
      h_bf[i] = f2bf(hnew);
      out_relu[row * H + j] = f2bf(fmaxf(hnew, 0.f));
    }
    __syncthreads();
  }
}

// ---------------------------------------------------------------------------
// len_info + mean over S + contrast (bf16 out). One WG per batch row; H==256.
// ---------------------------------------------------------------------------
__global__ __launch_bounds__(256) void k_mean_contrast(
    const float* __restrict__ text_fea,   // [B,S,256]
    const float* __restrict__ len_mask,   // [B,S]
    float* __restrict__ len_info,         // [B]
    unsigned short* __restrict__ con_bf,  // [B,S,256]
    int S)
{
  __shared__ float red[256];
  __shared__ float linv;
  int b = blockIdx.x;
  float lm = (threadIdx.x < (unsigned)S) ? len_mask[(size_t)b * S + threadIdx.x] : 0.f;
  red[threadIdx.x] = lm;
  __syncthreads();
  for (int s = 128; s > 0; s >>= 1) {
    if ((int)threadIdx.x < s) red[threadIdx.x] += red[threadIdx.x + s];
    __syncthreads();
  }
  if (threadIdx.x == 0) { len_info[b] = red[0]; linv = 1.f / red[0]; }
  __syncthreads();

  int c = threadIdx.x;                       // 256 channels
  const float* tf = text_fea + (size_t)b * S * 256;
  float sum = 0.f;
  for (int s = 0; s < S; ++s) sum += tf[(size_t)s * 256 + c];
  float mean = sum * linv;
  unsigned short* cb = con_bf + (size_t)b * S * 256;
  for (int s = 0; s < S; ++s) cb[(size_t)s * 256 + c] = f2bf(tf[(size_t)s * 256 + c] - mean);
}

// ---------------------------------------------------------------------------
// 3-token / 2-head attention core. One wave per position n; lane l owns dims
// [8l, 8l+8) so head0 = lanes 0-15, head1 = lanes 16-31 (xor-reduce over 16).
// qkv rows are [q(256) | k(256) | v(256)].
// ---------------------------------------------------------------------------
__global__ __launch_bounds__(256) void k_attn(
    const float* __restrict__ qkv_img,   // [B, 768]
    const float* __restrict__ qkv_text,  // [BS, 768]
    const float* __restrict__ qkv_con,   // [BS, 768]
    unsigned short* __restrict__ att,    // [3, BS, 256] bf16
    int BS, int S)
{
  int n = blockIdx.x * 8 + (threadIdx.x >> 5);
  int lane = threadIdx.x & 31;
  if (n >= BS) return;
  int b = n / S;
  const float* base0 = qkv_img  + (size_t)b * 768;
  const float* base1 = qkv_text + (size_t)n * 768;
  const float* base2 = qkv_con  + (size_t)n * 768;
  const float* base[3] = { base0, base1, base2 };

  const float scale = 0.08838834764831845f;   // 1/sqrt(128)
  float q[3][8], k[3][8], v[3][8];
  int d0 = lane * 8;
  for (int ti = 0; ti < 3; ++ti)
    for (int e = 0; e < 8; ++e) {
      q[ti][e] = base[ti][d0 + e] * scale;
      k[ti][e] = base[ti][256 + d0 + e];
      v[ti][e] = base[ti][512 + d0 + e];
    }

  float sc[3][3];
  for (int ti = 0; ti < 3; ++ti)
    for (int tj = 0; tj < 3; ++tj) {
      float p = 0.f;
      for (int e = 0; e < 8; ++e) p += q[ti][e] * k[tj][e];
      p += __shfl_xor(p, 1, 32);  p += __shfl_xor(p, 2, 32);
      p += __shfl_xor(p, 4, 32);  p += __shfl_xor(p, 8, 32);   // per-head dot
      sc[ti][tj] = p;
    }

  for (int ti = 0; ti < 3; ++ti) {
    float mx = fmaxf(sc[ti][0], fmaxf(sc[ti][1], sc[ti][2]));
    float e0 = __expf(sc[ti][0] - mx), e1 = __expf(sc[ti][1] - mx), e2 = __expf(sc[ti][2] - mx);
    float inv = 1.f / (e0 + e1 + e2);
    float a0 = e0 * inv, a1 = e1 * inv, a2 = e2 * inv;
    union { unsigned short s[8]; u32x4 u; } ou;
    for (int e = 0; e < 8; ++e)
      ou.s[e] = f2bf(a0 * v[0][e] + a1 * v[1][e] + a2 * v[2][e]);
    *(u32x4*)(att + ((size_t)ti * BS + n) * 256 + d0) = ou.u;
  }
}

// mean over the 3 tokens -> bf16
__global__ __launch_bounds__(256) void k_reduce3(const float* __restrict__ o,
                                                 unsigned short* __restrict__ last,
                                                 int n, int stride) {
  int i = blockIdx.x * 256 + threadIdx.x;
  if (i < n)
    last[i] = f2bf((o[i] + o[i + stride] + o[i + 2 * stride]) * (1.f / 3.f));
}

// ---------------------------------------------------------------------------
// Per-position masked-softmax CE loss. One wave per (b,s); lanes stripe F=190.
// ---------------------------------------------------------------------------
__global__ __launch_bounds__(256) void k_loss_pos(
    const float* __restrict__ logits, int ldo,
    const float* __restrict__ label_mask, const int* __restrict__ labels,
    float* __restrict__ loss_el, int BS, int F)
{
  int pos = blockIdx.x * 8 + (threadIdx.x >> 5);
  int lane = threadIdx.x & 31;
  if (pos >= BS) return;
  const float* l = logits + (size_t)pos * ldo;
  float lv[6], mx = -1e30f;
  for (int i = 0; i < 6; ++i) {
    int f = lane + 32 * i;
    lv[i] = (f < F) ? l[f] : -1e30f;
    mx = fmaxf(mx, lv[i]);
  }
  for (int m = 16; m >= 1; m >>= 1) mx = fmaxf(mx, __shfl_xor(mx, m, 32));
  float se = 0.f;
  for (int i = 0; i < 6; ++i) { int f = lane + 32 * i; if (f < F) se += __expf(lv[i] - mx); }
  for (int m = 16; m >= 1; m >>= 1) se += __shfl_xor(se, m, 32);
  float lse = mx + __logf(se);
  int lab = labels[pos];
  const float* lm = label_mask + (size_t)pos * F;
  float acc = 0.f;
  for (int i = 0; i < 6; ++i) {
    int f = lane + 32 * i;
    if (f < F) {
      float oh = fmaxf(lm[f], (f == lab) ? 1.f : 0.f);
      acc += oh * (lse - lv[i]);
    }
  }
  for (int m = 16; m >= 1; m >>= 1) acc += __shfl_xor(acc, m, 32);
  if (lane == 0) loss_el[pos] = acc;
}

__global__ __launch_bounds__(256) void k_loss_final(
    const float* __restrict__ loss_el, const float* __restrict__ len_mask,
    const float* __restrict__ len_info, float* __restrict__ out, int B, int S)
{
  __shared__ float red[256];
  float acc = 0.f;
  if ((int)threadIdx.x < B) {
    int b = threadIdx.x;
    float s = 0.f;
    for (int t = 0; t < S; ++t) s += loss_el[(size_t)b * S + t] * len_mask[(size_t)b * S + t];
    acc = s / len_info[b];
  }
  red[threadIdx.x] = acc;
  __syncthreads();
  for (int s = 128; s > 0; s >>= 1) {
    if ((int)threadIdx.x < s) red[threadIdx.x] += red[threadIdx.x + s];
    __syncthreads();
  }
  if (threadIdx.x == 0) out[0] = red[0] / (float)B;
}

// ---------------------------------------------------------------------------
// Host-side orchestration
// ---------------------------------------------------------------------------
extern "C" void kernel_launch(void* const* d_in, const int* in_sizes, int n_in,
                              void* d_out, int out_size, void* d_ws, size_t ws_size,
                              hipStream_t stream) {
  (void)in_sizes; (void)n_in; (void)out_size; (void)ws_size;
  const int B = 128, S = 128, D = 2048, H = 256, H1 = 128, F = 190, Fp = 192;
  const int BS = B * S;

  const float* img_emb   = (const float*)d_in[0];
  const float* text_embs = (const float*)d_in[1];
  const float* len_mask  = (const float*)d_in[2];
  const float* label_msk = (const float*)d_in[3];
  const int*   labels    = (const int*)d_in[4];
  const float* img_W  = (const float*)d_in[5];
  const float* img_b  = (const float*)d_in[6];
  const float* img_g  = (const float*)d_in[7];
  const float* img_be = (const float*)d_in[8];
  const float* img_m  = (const float*)d_in[9];
  const float* img_v  = (const float*)d_in[10];
  const float* g1_Wih = (const float*)d_in[11];
  const float* g1_Whh = (const float*)d_in[12];
  const float* g1_bih = (const float*)d_in[13];
  const float* g1_bhh = (const float*)d_in[14];
  const float* tl_W   = (const float*)d_in[15];
  const float* tl_b   = (const float*)d_in[16];
  const float* t_g    = (const float*)d_in[17];
  const float* t_be   = (const float*)d_in[18];
  const float* t_m    = (const float*)d_in[19];
  const float* t_v    = (const float*)d_in[20];
  const float* g2_Wih = (const float*)d_in[21];
  const float* g2_Whh = (const float*)d_in[22];
  const float* g2_bih = (const float*)d_in[23];
  const float* g2_bhh = (const float*)d_in[24];
  const float* cl_W   = (const float*)d_in[25];
  const float* cl_b   = (const float*)d_in[26];
  const float* c_g    = (const float*)d_in[27];
  const float* c_be   = (const float*)d_in[28];
  const float* c_m    = (const float*)d_in[29];
  const float* c_v    = (const float*)d_in[30];
  const float* attn_Win  = (const float*)d_in[31];
  const float* attn_bin  = (const float*)d_in[32];
  const float* attn_Wout = (const float*)d_in[33];
  const float* attn_bout = (const float*)d_in[34];
  const float* pred_W = (const float*)d_in[35];
  const float* pred_b = (const float*)d_in[36];

  char* ws = (char*)d_ws;
  size_t off = 0;
  auto alloc = [&](size_t bytes) -> void* {
    off = (off + 255) & ~(size_t)255;
    void* p = ws + off;
    off += bytes;
    return p;
  };
  typedef unsigned short u16;

  // persistent
  u16* w_img  = (u16*)alloc((size_t)H * D * 2);
  u16* w_g1ih = (u16*)alloc((size_t)3 * H1 * D * 2);
  u16* w_g1hh = (u16*)alloc((size_t)3 * H1 * H1 * 2);
  u16* w_tl   = (u16*)alloc((size_t)H * H1 * 2);
  u16* w_g2ih = (u16*)alloc((size_t)3 * H * H * 2);
  u16* w_g2hh = (u16*)alloc((size_t)3 * H * H * 2);
  u16* w_cl   = (u16*)alloc((size_t)H * H * 2);
  u16* w_ain  = (u16*)alloc((size_t)3 * H * H * 2);
  u16* w_aout = (u16*)alloc((size_t)H * H * 2);
  u16* w_pred = (u16*)alloc((size_t)Fp * H * 2);
  u16* imge_bf   = (u16*)alloc((size_t)B * D * 2);
  u16* imgfea_bf = (u16*)alloc((size_t)B * H * 2);
  u16* t_relu    = (u16*)alloc((size_t)BS * H1 * 2);
  float* text_f  = (float*)alloc((size_t)BS * H * 4);
  u16* text_b    = (u16*)alloc((size_t)BS * H * 2);
  float* len_inf = (float*)alloc((size_t)B * 4);
  u16* con_bf    = (u16*)alloc((size_t)BS * H * 2);
  u16* c_relu    = (u16*)alloc((size_t)BS * H * 2);
  u16* cfea_bf   = (u16*)alloc((size_t)BS * H * 2);
  float* qkvi_f  = (float*)alloc((size_t)B * 3 * H * 4);
  u16* last_bf   = (u16*)alloc((size_t)BS * H * 2);
  float* loss_el = (float*)alloc((size_t)BS * 4);

  // aliased regions (non-overlapping lifetimes)
  char* RA = (char*)alloc((size_t)BS * D * 2);      // text_bf -> qkv_text_f
  char* RB = (char*)alloc((size_t)BS * 384 * 4);    // xp1 -> att_bf (equal size)
  char* RC = (char*)alloc((size_t)BS * 768 * 4);    // xp2 -> o_proj (equal size)
  char* RD = (char*)alloc((size_t)BS * 768 * 4);    // qkv_con_f -> logits

  u16*   text_bf  = (u16*)RA;
  float* qkvt_f   = (float*)RA;
  float* xp1      = (float*)RB;
  u16*   att_bf   = (u16*)RB;
  float* xp2      = (float*)RC;
  float* o_proj   = (float*)RC;
  float* qkvc_f   = (float*)RD;
  float* logits   = (float*)RD;

  auto conv = [&](const float* s, u16* d, int n) {
    k_convert<<<(n + 255) / 256, 256, 0, stream>>>(s, d, n);
  };
  auto gemm = [&](const u16* A, int lda, const u16* W, int ldw,
                  const float* bias, const float* g, const float* be,
                  const float* mu, const float* va,
                  float* oF, u16* oB, int ldo, int M, int Np, int K,
                  int nst, int relu) {
    dim3 gr((Np + 127) / 128, (M + 63) / 64);
    k_gemm_bf16<<<gr, 256, 0, stream>>>(A, lda, W, ldw, bias, g, be, mu, va,
                                        oF, oB, ldo, M, Np, K, nst, relu);
  };

  // 0) weight / input conversions to bf16
  conv(img_W,  w_img,  H * D);
  conv(g1_Wih, w_g1ih, 3 * H1 * D);
  conv(g1_Whh, w_g1hh, 3 * H1 * H1);
  conv(tl_W,   w_tl,   H * H1);
  conv(g2_Wih, w_g2ih, 3 * H * H);
  conv(g2_Whh, w_g2hh, 3 * H * H);
  conv(cl_W,   w_cl,   H * H);
  conv(attn_Win,  w_ain,  3 * H * H);
  conv(attn_Wout, w_aout, H * H);
  k_convert_pad<<<(Fp * H + 255) / 256, 256, 0, stream>>>(pred_W, w_pred, F, H, Fp);
  conv(img_emb,   imge_bf, B * D);
  conv(text_embs, text_bf, BS * D);

  // 1) img encoder: Linear + BN + ReLU -> bf16 token
  gemm(imge_bf, D, w_img, D, img_b, img_g, img_be, img_m, img_v,
       nullptr, imgfea_bf, H, B, H, D, H, 1);

  // 2) GRU1 input GEMM (the big one): xp1 = text @ g1_Wih^T + bih
  gemm(text_bf, D, w_g1ih, D, g1_bih, nullptr, nullptr, nullptr, nullptr,
       xp1, nullptr, 3 * H1, BS, 3 * H1, D, 3 * H1, 0);

  // 3) GRU1 recurrence (batch split across 8 WGPs), emits relu(h) bf16
  {
    size_t smem = (size_t)16 * H1 * 2 + (size_t)16 * H1 * 4 + (size_t)16 * 3 * H1 * 4;
    k_gru<<<B / 16, 256, smem, stream>>>(xp1, w_g1hh, g1_bhh, t_relu, H1, S);
  }

  // 4) tl Linear + BN + ReLU -> text_fea (f32 for mean/contrast, bf16 token)
  gemm(t_relu, H1, w_tl, H1, tl_b, t_g, t_be, t_m, t_v,
       text_f, text_b, H, BS, H, H1, H, 1);

  // 5) len_info, mean over S, contrast (bf16)
  k_mean_contrast<<<B, 256, 0, stream>>>(text_f, len_mask, len_inf, con_bf, S);

  // 6) GRU2 input GEMM
  gemm(con_bf, H, w_g2ih, H, g2_bih, nullptr, nullptr, nullptr, nullptr,
       xp2, nullptr, 3 * H, BS, 3 * H, H, 3 * H, 0);

  // 7) GRU2 recurrence
  {
    size_t smem = (size_t)16 * H * 2 + (size_t)16 * H * 4 + (size_t)16 * 3 * H * 4;
    k_gru<<<B / 16, 256, smem, stream>>>(xp2, w_g2hh, g2_bhh, c_relu, H, S);
  }

  // 8) cl Linear + BN + ReLU -> contrast_fea bf16 token
  gemm(c_relu, H, w_cl, H, cl_b, c_g, c_be, c_m, c_v,
       nullptr, cfea_bf, H, BS, H, H, H, 1);

  // 9) fused QKV projections (Win = [Wq;Wk;Wv], N = 768) per token
  gemm(imgfea_bf, H, w_ain, H, attn_bin, nullptr, nullptr, nullptr, nullptr,
       qkvi_f, nullptr, 3 * H, B, 3 * H, H, 3 * H, 0);
  gemm(text_b, H, w_ain, H, attn_bin, nullptr, nullptr, nullptr, nullptr,
       qkvt_f, nullptr, 3 * H, BS, 3 * H, H, 3 * H, 0);
  gemm(cfea_bf, H, w_ain, H, attn_bin, nullptr, nullptr, nullptr, nullptr,
       qkvc_f, nullptr, 3 * H, BS, 3 * H, H, 3 * H, 0);

  // 10) 3-token 2-head attention core -> att_bf [3, BS, 256]
  k_attn<<<BS / 8, 256, 0, stream>>>(qkvi_f, qkvt_f, qkvc_f, att_bf, BS, S);

  // 11) output projection over all 3*BS rows
  gemm(att_bf, H, w_aout, H, attn_bout, nullptr, nullptr, nullptr, nullptr,
       o_proj, nullptr, H, 3 * BS, H, H, H, 0);

  // 12) mean over the 3 tokens
  k_reduce3<<<(BS * H + 255) / 256, 256, 0, stream>>>(o_proj, last_bf, BS * H, BS * H);

  // 13) classifier GEMM (W zero-padded 190 -> 192 rows; store 190 cols)
  gemm(last_bf, H, w_pred, H, pred_b, nullptr, nullptr, nullptr, nullptr,
       logits, nullptr, Fp, BS, Fp, H, F, 0);

  // 14) masked CE loss
  k_loss_pos<<<BS / 8, 256, 0, stream>>>(logits, Fp, label_msk, labels, loss_el, BS, F);
  k_loss_final<<<1, 256, 0, stream>>>(loss_el, len_mask, len_inf, (float*)d_out, B, S);
}
